// DirectionalProcessor_15753940042279
// MI455X (gfx1250) — matrix operand order repaired and checked by reading, not compiled, not run on hardware
//
#include <hip/hip_runtime.h>
#include <hip/hip_bf16.h>

typedef __attribute__((ext_vector_type(4)))  __bf16 v4bf;
typedef __attribute__((ext_vector_type(8)))  __bf16 v8bf;
typedef __attribute__((ext_vector_type(16))) __bf16 v16bf;
typedef __attribute__((ext_vector_type(4)))  float  v4f;
typedef __attribute__((ext_vector_type(8)))  float  v8f;

#define CCH 256
#define HH  64
#define WW  64
#define BBN 16

// ---------------------------------------------------------------------------
// Kernel 1: fold the two matmuls.
// WeffT[d][e][c] = sum_k Wd[d][c][k] * Wc[e][d*256 + k], split into bf16 hi/lo.
// Stored transposed (e-major, c contiguous) so WMMA B-fragments are contiguous
// along K. All of Wd/Wc (4 MB) is L2-resident; one-shot ~µs cost.
// ---------------------------------------------------------------------------
__global__ __launch_bounds__(256) void weff_fold_kernel(
    const float* __restrict__ Wd, const float* __restrict__ Wc,
    __bf16* __restrict__ whi, __bf16* __restrict__ wlo) {
  int gid = blockIdx.x * 256 + threadIdx.x;   // 8 * 256 * 64 threads
  int cg = gid & 63;                          // c-group (4 c's)
  int e  = (gid >> 6) & 255;                  // output channel
  int d  = gid >> 14;                         // direction
  int c0 = cg * 4;

  const float* wcrow = Wc + (size_t)e * (8 * CCH) + d * CCH;   // contiguous in k
  const float* wdbas = Wd + ((size_t)d * CCH + c0) * CCH;      // 4 rows, contiguous in k

  float a0 = 0.f, a1 = 0.f, a2 = 0.f, a3 = 0.f;
#pragma unroll 4
  for (int k = 0; k < CCH; ++k) {
    float wc = wcrow[k];
    a0 += wdbas[0 * CCH + k] * wc;
    a1 += wdbas[1 * CCH + k] * wc;
    a2 += wdbas[2 * CCH + k] * wc;
    a3 += wdbas[3 * CCH + k] * wc;
  }

  float acc[4] = {a0, a1, a2, a3};
  size_t o = ((size_t)d * CCH + e) * CCH + c0;  // WeffT[d][e][c0..c0+3]
#pragma unroll
  for (int j = 0; j < 4; ++j) {
    float v = acc[j];
    __bf16 h = (__bf16)v;                       // hi part (RNE)
    __bf16 l = (__bf16)(v - (float)h);          // residual -> lo part
    whi[o + j] = h;
    wlo[o + j] = l;
  }
}

// ---------------------------------------------------------------------------
// Kernel 2: 8-direction shifted GEMM via bf16x3 WMMA (fp32-class accuracy).
// One workgroup per image row (b, y): 8 waves of 32.
// Wave w owns ALL FOUR 16-pixel M-tiles and TWO 16-channel N-tiles
// (nt = 2w, 2w+1): B fragments (global/L2) are loaded once per (d, k-step)
// and amortized over 4 M-tiles; A fragments come from conflict-free LDS b128
// loads (row stride 40 bf16 = 20 banks).
// Per 32-channel K chunk: stage the 3x66-pixel halo slab in LDS (bf16 hi/lo),
// then all 8 directions consume it with shifted addressing.
// ---------------------------------------------------------------------------
__global__ __launch_bounds__(256) void dirconv_wmma_kernel(
    const float* __restrict__ g,
    const __bf16* __restrict__ whi, const __bf16* __restrict__ wlo,
    const float* __restrict__ bc, float* __restrict__ out) {

  __shared__ __bf16 sHi[198 * 40];   // [3 rows * 66 px][40 (32 ch + pad)]
  __shared__ __bf16 sLo[198 * 40];

  const int tid   = threadIdx.x;
  const int lane  = tid & 31;
  const int wave  = tid >> 5;
  const int nt0   = wave * 2;        // this wave's two 16-wide N tiles
  const int b     = blockIdx.x >> 6;
  const int y     = blockIdx.x & 63;
  const int ln16  = lane & 15;
  const bool hi16 = lane >= 16;

  // (dx, dy) per reference DIRECTIONS; A row for output (y,x) is g[y-dy][x-dx]
  constexpr int dirx[8] = {0, 1, 1, 1, 0, -1, -1, -1};
  constexpr int diry[8] = {-1, -1, 0, 1, 1, 1, 0, -1};

  v8f acc[8];                        // [mtile][ntile j] -> acc[mt*2+j]
#pragma unroll
  for (int i = 0; i < 8; ++i)
#pragma unroll
    for (int j = 0; j < 8; ++j) acc[i][j] = 0.0f;

  // ISA 7.12.2 fragment K-chunks (16-bit, 16x16x32):
  //   A: lanes 0-15: K in [0,8) u [16,24); lanes 16-31: K in [8,16) u [24,32)
  //   B: lanes 0-15: K in [0,8) u [16,24); lanes 16-31: K in [8,16) u [24,32)
  const int k0a = hi16 ? 8 : 0;
  const int k0b = hi16 ? 8 : 0;
  const int e0  = nt0 * 16 + ln16;   // lane's N (output channel) for j=0

  for (int kk = 0; kk < CCH; kk += 32) {
    __syncthreads();  // previous chunk's LDS reads must finish before restaging

    // ---- stage halo slab: rows y-1..y+1, x -1..64, channels kk..kk+31 ----
    for (int i = tid; i < 3 * 66 * 8; i += 256) {
      int ry  = i / (66 * 8);
      int rem = i - ry * (66 * 8);
      int px  = rem >> 3;
      int cq  = rem & 7;
      int yy  = y + ry - 1;
      int xx  = px - 1;
      v4f v = {0.f, 0.f, 0.f, 0.f};
      if (yy >= 0 && yy < HH && xx >= 0 && xx < WW)
        v = *(const v4f*)(g + ((((size_t)b * HH + yy) * WW + xx) * CCH + kk + cq * 4));
      v4bf h, l;
#pragma unroll
      for (int j = 0; j < 4; ++j) {
        __bf16 hh = (__bf16)v[j];
        h[j] = hh;
        l[j] = (__bf16)(v[j] - (float)hh);
      }
      int base = (ry * 66 + px) * 40 + cq * 4;
      *(v4bf*)(sHi + base) = h;
      *(v4bf*)(sLo + base) = l;
    }
    __syncthreads();

    // ---- 8 directions consume the slab ----
#pragma unroll
    for (int d = 0; d < 8; ++d) {
      // B fragments for this wave's two N tiles (hoisted over all 4 M tiles)
      v16bf bHi[2], bLo[2];
#pragma unroll
      for (int j = 0; j < 2; ++j) {
        size_t boff = ((size_t)d * CCH + (e0 + j * 16)) * CCH + kk + k0b;
        v8bf b0 = *(const v8bf*)(whi + boff);
        v8bf b1 = *(const v8bf*)(whi + boff + 16);
        v8bf b2 = *(const v8bf*)(wlo + boff);
        v8bf b3 = *(const v8bf*)(wlo + boff + 16);
        bHi[j] = __builtin_shufflevector(b0, b1, 0,1,2,3,4,5,6,7,8,9,10,11,12,13,14,15);
        bLo[j] = __builtin_shufflevector(b2, b3, 0,1,2,3,4,5,6,7,8,9,10,11,12,13,14,15);
      }

      const int ry   = 1 - diry[d];                 // yy = y - dy
      const int pxb  = ln16 - dirx[d] + 1;          // xx = x - dx

#pragma unroll
      for (int mt = 0; mt < 4; ++mt) {
        const int arow = (ry * 66 + mt * 16 + pxb) * 40;
        v8bf a0 = *(const v8bf*)(sHi + arow + k0a);
        v8bf a1 = *(const v8bf*)(sHi + arow + k0a + 16);
        v8bf a2 = *(const v8bf*)(sLo + arow + k0a);
        v8bf a3 = *(const v8bf*)(sLo + arow + k0a + 16);
        v16bf aHi = __builtin_shufflevector(a0, a1, 0,1,2,3,4,5,6,7,8,9,10,11,12,13,14,15);
        v16bf aLo = __builtin_shufflevector(a2, a3, 0,1,2,3,4,5,6,7,8,9,10,11,12,13,14,15);

#pragma unroll
        for (int j = 0; j < 2; ++j) {
          // bf16x3 split-precision accumulate (fp32 C/D)
          acc[mt * 2 + j] = __builtin_amdgcn_wmma_f32_16x16x32_bf16(
              false, aHi, false, bHi[j], (short)0, acc[mt * 2 + j], false, false);
          acc[mt * 2 + j] = __builtin_amdgcn_wmma_f32_16x16x32_bf16(
              false, aLo, false, bHi[j], (short)0, acc[mt * 2 + j], false, false);
          acc[mt * 2 + j] = __builtin_amdgcn_wmma_f32_16x16x32_bf16(
              false, aHi, false, bLo[j], (short)0, acc[mt * 2 + j], false, false);
        }
      }
    }
  }

  // ---- epilogue: bias + coalesced fp32 stores ----
  // C/D layout: VGPR r, lanes 0-15 -> (M=r, N=lane); lanes 16-31 -> (M=r+8, N=lane-16)
  const int m0 = hi16 ? 8 : 0;
  float bias0 = bc[e0];
  float bias1 = bc[e0 + 16];
#pragma unroll
  for (int mt = 0; mt < 4; ++mt) {
#pragma unroll
    for (int j = 0; j < 2; ++j) {
      int e = e0 + j * 16;
      float bias = j ? bias1 : bias0;
#pragma unroll
      for (int r = 0; r < 8; ++r) {
        int x = mt * 16 + r + m0;
        out[(((size_t)b * HH + y) * WW + x) * CCH + e] = acc[mt * 2 + j][r] + bias;
      }
    }
  }
}

// ---------------------------------------------------------------------------
extern "C" void kernel_launch(void* const* d_in, const int* in_sizes, int n_in,
                              void* d_out, int out_size, void* d_ws, size_t ws_size,
                              hipStream_t stream) {
  const float* g  = (const float*)d_in[0];  // [16,64,64,256]
  const float* Wd = (const float*)d_in[1];  // [8,256,256]
  const float* Wc = (const float*)d_in[2];  // [256, 8*256]
  const float* bc = (const float*)d_in[3];  // [256]
  float* out = (float*)d_out;               // [16,64,64,256]

  __bf16* whi = (__bf16*)d_ws;                       // 8*256*256 bf16 = 1 MB
  __bf16* wlo = whi + (size_t)8 * CCH * CCH;         // +1 MB

  // Fold Wd x Wc -> WeffT (bf16 hi/lo), then the shifted WMMA GEMM.
  weff_fold_kernel<<<512, 256, 0, stream>>>(Wd, Wc, whi, wlo);
  dirconv_wmma_kernel<<<BBN * HH, 256, 0, stream>>>(g, whi, wlo, bc, out);
}
